// MoE_6554120094085
// MI455X (gfx1250) — compile-verified, matrix operand order checked by reading, not run on hardware
//
#include <hip/hip_runtime.h>
#include <stdint.h>

#define EXPERTS 8
#define DD      1024
#define BM      128
#define BN      128
#define BK      32
#define LDA     40              // padded bf16 row stride (80B = 5*16B, async-friendly)
#define KT      (DD / BK)       // 32 K-steps
#define SZP     (BM * LDA * 2)  // bytes per plane stage (A and B identical: 10,240 B)

typedef __attribute__((ext_vector_type(16))) __bf16 v16bf;
typedef __attribute__((ext_vector_type(8)))  float  v8f;

// ---------------------------------------------------------------- helpers ---

__device__ __forceinline__ v8f wmma_bf16(v16bf a, v16bf b, v8f c) {
  return __builtin_amdgcn_wmma_f32_16x16x32_bf16(false, a, false, b, (short)0, c,
                                                 false, false);
}

// generic -> LDS byte offset (addrspacecast strips the aperture)
typedef __attribute__((address_space(3))) const void lds_cvoid;
__device__ __forceinline__ uint32_t lds_addr32(const void* p) {
  return (uint32_t)(uintptr_t)(lds_cvoid*)p;
}

// CDNA5 async copy: 16B per lane, global -> LDS, tracked by ASYNCcnt
__device__ __forceinline__ void async_b128(uint32_t lds_off, const void* gaddr) {
  asm volatile("global_load_async_to_lds_b128 %0, %1, off"
               :: "v"(lds_off), "v"(gaddr) : "memory");
}
__device__ __forceinline__ void wait_async0() {
  asm volatile("s_wait_asynccnt 0x0" ::: "memory");
}

// Fragment load: CDNA5 16-bit operand layout. lanes 0-15: row=lane, K{0..7,16..23};
// lanes 16-31: row=lane-16, K{8..15,24..31}.
__device__ __forceinline__ v16bf lds_frag(const __bf16* plane, int row, int half) {
  union { uint4 q[2]; v16bf v; } u;
  const __bf16* p = plane + row * LDA + half * 8;
  u.q[0] = *(const uint4*)(p);
  u.q[1] = *(const uint4*)(p + 16);
  return u.v;
}

__device__ __forceinline__ void cvt_hilo(float x, __bf16& h, __bf16& l) {
  h = (__bf16)x;
  l = (__bf16)(x - (float)h);
}

// ----------------------------------------------------- one-time split kernels

__global__ __launch_bounds__(256) void split_plain(
    const float* __restrict__ src, __bf16* __restrict__ hi,
    __bf16* __restrict__ lo, int n)
{
  const int i = (blockIdx.x * 256 + threadIdx.x) * 4;
  if (i >= n) return;
  const float4 f = *(const float4*)(src + i);
  union { __bf16 h[4]; uint2 u; } H, L;
  cvt_hilo(f.x, H.h[0], L.h[0]);
  cvt_hilo(f.y, H.h[1], L.h[1]);
  cvt_hilo(f.z, H.h[2], L.h[2]);
  cvt_hilo(f.w, H.h[3], L.h[3]);
  *(uint2*)(hi + i) = H.u;
  *(uint2*)(lo + i) = L.u;
}

// transpose + split: W[e][k][n] f32 -> WT hi/lo [e][n][k] bf16 (32x32 LDS tiles)
__global__ __launch_bounds__(256) void split_transpose(
    const float* __restrict__ W, __bf16* __restrict__ hiT,
    __bf16* __restrict__ loT)
{
  __shared__ float tile[32][33];
  const int e = blockIdx.z;
  const int k0 = blockIdx.y * 32, n0 = blockIdx.x * 32;
  const int tn = threadIdx.x & 31;
  const int g4 = threadIdx.x >> 5;
  const size_t eo = (size_t)e << 20;
  const float* src = W + eo + (size_t)k0 * DD + n0;
#pragma unroll
  for (int j = 0; j < 4; ++j) {
    const int k = g4 * 4 + j;
    tile[k][tn] = src[(size_t)k * DD + tn];
  }
  __syncthreads();
#pragma unroll
  for (int j = 0; j < 4; ++j) {
    const int nn = g4 * 4 + j;
    __bf16 h, l;
    cvt_hilo(tile[tn][nn], h, l);
    const size_t o = eo + (size_t)(n0 + nn) * DD + k0 + tn;
    hiT[o] = h;
    loT[o] = l;
  }
}

// ------------------------------------------------------------ gate kernel ---

__global__ __launch_bounds__(256) void gate_topk(
    const float* __restrict__ x, const float* __restrict__ Wg,
    const float* __restrict__ bg, float* __restrict__ gate_out,
    unsigned* __restrict__ cnt, unsigned* __restrict__ lists,
    float* __restrict__ wlist, int Ntok)
{
  __shared__ float wgs[DD * EXPERTS];
  for (int i = threadIdx.x; i < DD * EXPERTS; i += 256) wgs[i] = Wg[i];
  __syncthreads();

  const int wave = threadIdx.x >> 5;
  const int lane = threadIdx.x & 31;
  const int n = blockIdx.x * 8 + wave;
  if (n >= Ntok) return;

  const float* xr = x + (size_t)n * DD;
  float acc[EXPERTS];
#pragma unroll
  for (int e = 0; e < EXPERTS; ++e) acc[e] = 0.f;
  for (int d = lane; d < DD; d += 32) {
    const float xv = xr[d];
#pragma unroll
    for (int e = 0; e < EXPERTS; ++e)
      acc[e] = fmaf(xv, wgs[d * EXPERTS + e], acc[e]);
  }
#pragma unroll
  for (int e = 0; e < EXPERTS; ++e) {
    float v = acc[e];
#pragma unroll
    for (int m = 16; m >= 1; m >>= 1) v += __shfl_xor(v, m, 32);
    acc[e] = v;
  }

  if (lane == 0) {
    float lg[EXPERTS];
    float mx = acc[0] + bg[0];
#pragma unroll
    for (int e = 0; e < EXPERTS; ++e) { lg[e] = acc[e] + bg[e]; mx = fmaxf(mx, lg[e]); }
    float p[EXPERTS], s = 0.f;
#pragma unroll
    for (int e = 0; e < EXPERTS; ++e) { p[e] = __expf(lg[e] - mx); s += p[e]; }
    const float inv = 1.f / s;
#pragma unroll
    for (int e = 0; e < EXPERTS; ++e) {
      p[e] *= inv;
      gate_out[(size_t)n * EXPERTS + e] = p[e];
    }
    int i0 = 0; float p0 = p[0];
#pragma unroll
    for (int e = 1; e < EXPERTS; ++e) if (p[e] > p0) { p0 = p[e]; i0 = e; }
    int i1 = (i0 == 0) ? 1 : 0; float p1 = p[i1];
#pragma unroll
    for (int e = 0; e < EXPERTS; ++e)
      if (e != i0 && p[e] > p1) { p1 = p[e]; i1 = e; }
    const float t  = __expf(p1 - p0);
    const float w0 = 1.f / (1.f + t);
    const float w1 = t * w0;

    const unsigned s0 = atomicAdd(&cnt[i0], 1u);
    lists[(size_t)i0 * Ntok + s0] = (unsigned)(2 * n);
    wlist[(size_t)i0 * Ntok + s0] = w0;
    const unsigned s1 = atomicAdd(&cnt[i1], 1u);
    lists[(size_t)i1 * Ntok + s1] = (unsigned)(2 * n + 1);
    wlist[(size_t)i1 * Ntok + s1] = w1;
  }
}

// ----------------------------------------------------------- GEMM kernels ---
// Block tile 128x128, 8 waves (4Mx2N), wave tile 32x64 (2x4 WMMA tiles).
// acc += Ah*Bh + Ah*Bl + Al*Bh  (fp32-emulation via bf16 split).
// Staging = pure async copy; per K-step per wave: 24 ds_load_b128 : 24 WMMA.

template <int LAYER>
__global__ __launch_bounds__(256) void moe_gemm(
    const __bf16* __restrict__ Ah, const __bf16* __restrict__ Al,
    const __bf16* __restrict__ BhT, const __bf16* __restrict__ BlT,
    const float* __restrict__ bias, const unsigned* __restrict__ cnt,
    const unsigned* __restrict__ lists, const float* __restrict__ wlist,
    __bf16* __restrict__ hH, __bf16* __restrict__ hL,
    float* __restrict__ y, int Ntok)
{
  __shared__ __bf16 sAh[2][BM * LDA];
  __shared__ __bf16 sAl[2][BM * LDA];
  __shared__ __bf16 sBh[2][BN * LDA];
  __shared__ __bf16 sBl[2][BN * LDA];   // 81,920 B total

  const int e = blockIdx.z;
  const unsigned count = cnt[e];
  const int mBase = blockIdx.y * BM;
  if ((unsigned)mBase >= count) return;
  const int nBase = blockIdx.x * BN;
  const int t = threadIdx.x;
  const size_t eList = (size_t)e * Ntok;

  // --- per-thread async staging: rows r and r+64 of both A and B planes,
  //     16B chunk ch (4 chunks cover BK=32 bf16) ---
  const int r  = t >> 2;                 // 0..63
  const int ch = t & 3;
  const int gk = ch * 8;                 // bf16 elem offset within k-chunk

  const int slot0 = mBase + r, slot1 = mBase + r + 64;
  const unsigned val0 = (slot0 < (int)count) ? lists[eList + slot0] : lists[eList];
  const unsigned val1 = (slot1 < (int)count) ? lists[eList + slot1] : lists[eList];
  const size_t idx0 = (LAYER == 1) ? (size_t)(val0 >> 1) : (size_t)val0;
  const size_t idx1 = (LAYER == 1) ? (size_t)(val1 >> 1) : (size_t)val1;

  const __bf16* a0h = Ah + idx0 * DD + gk;
  const __bf16* a0l = Al + idx0 * DD + gk;
  const __bf16* a1h = Ah + idx1 * DD + gk;
  const __bf16* a1l = Al + idx1 * DD + gk;
  const size_t eo = (size_t)e << 20;
  const size_t bo0 = eo + (size_t)(nBase + r)      * DD + gk;
  const size_t bo1 = eo + (size_t)(nBase + r + 64) * DD + gk;
  const __bf16* b0h = BhT + bo0;
  const __bf16* b0l = BlT + bo0;
  const __bf16* b1h = BhT + bo1;
  const __bf16* b1l = BlT + bo1;

  const uint32_t o0 = (uint32_t)((r)      * LDA + gk) * 2;
  const uint32_t o1 = (uint32_t)((r + 64) * LDA + gk) * 2;
  const uint32_t bAh = lds_addr32(&sAh[0][0]);
  const uint32_t bAl = lds_addr32(&sAl[0][0]);
  const uint32_t bBh = lds_addr32(&sBh[0][0]);
  const uint32_t bBl = lds_addr32(&sBl[0][0]);

  auto stage_async = [&](int st, int kOff) {
    const uint32_t so = (uint32_t)st * SZP;
    async_b128(bAh + so + o0, a0h + kOff);
    async_b128(bAl + so + o0, a0l + kOff);
    async_b128(bAh + so + o1, a1h + kOff);
    async_b128(bAl + so + o1, a1l + kOff);
    async_b128(bBh + so + o0, b0h + kOff);
    async_b128(bBl + so + o0, b0l + kOff);
    async_b128(bBh + so + o1, b1h + kOff);
    async_b128(bBl + so + o1, b1l + kOff);
  };

  // --- wave tiling: 8 waves (4Mx2N), wave tile 32M x 64N ---
  const int wave = t >> 5, lane = t & 31;
  const int wm = (wave >> 1) * 32;        // 0,32,64,96
  const int wn = (wave & 1) * 64;         // 0,64
  const int lr = lane & 15;
  const int lh = lane >> 4;

  v8f acc[2][4];
#pragma unroll
  for (int mi = 0; mi < 2; ++mi)
#pragma unroll
    for (int ni = 0; ni < 4; ++ni)
      acc[mi][ni] = (v8f){0.f, 0.f, 0.f, 0.f, 0.f, 0.f, 0.f, 0.f};

  stage_async(0, 0);                       // prologue

  for (int kt = 0; kt < KT; ++kt) {
    wait_async0();                         // my stage-kt loads done
    __syncthreads();                       // everyone's done; prev readers done
    const int st = kt & 1;
    if (kt + 1 < KT) stage_async(st ^ 1, (kt + 1) * BK);   // overlap compute

    v16bf aH[2], aL[2], bH[4], bL[4];
#pragma unroll
    for (int i = 0; i < 2; ++i) {
      aH[i] = lds_frag(sAh[st], wm + i * 16 + lr, lh);
      aL[i] = lds_frag(sAl[st], wm + i * 16 + lr, lh);
    }
#pragma unroll
    for (int i = 0; i < 4; ++i) {
      bH[i] = lds_frag(sBh[st], wn + i * 16 + lr, lh);
      bL[i] = lds_frag(sBl[st], wn + i * 16 + lr, lh);
    }
#pragma unroll
    for (int mi = 0; mi < 2; ++mi)
#pragma unroll
      for (int ni = 0; ni < 4; ++ni) {
        v8f c = acc[mi][ni];
        c = wmma_bf16(aH[mi], bH[ni], c);
        c = wmma_bf16(aH[mi], bL[ni], c);
        c = wmma_bf16(aL[mi], bH[ni], c);
        acc[mi][ni] = c;
      }
  }

  // --- epilogue: C layout lanes0-15 -> M=j, lanes16-31 -> M=j+8, N=lane&15 ---
#pragma unroll
  for (int mi = 0; mi < 2; ++mi) {
#pragma unroll
    for (int ni = 0; ni < 4; ++ni) {
      const v8f c = acc[mi][ni];
      const int gcol = nBase + wn + ni * 16 + lr;
      const float bb = bias[(size_t)e * DD + gcol];
#pragma unroll
      for (int j = 0; j < 8; ++j) {
        const int slot = mBase + wm + mi * 16 + lh * 8 + j;
        if (slot < (int)count) {
          const unsigned val = lists[eList + slot];
          float v = c[j] + bb;
          if (LAYER == 1) {
            v = fmaxf(v, 0.f);
            __bf16 h, l;
            cvt_hilo(v, h, l);
            hH[(size_t)val * DD + gcol] = h;
            hL[(size_t)val * DD + gcol] = l;
          } else {
            const float w = wlist[eList + slot];
            __hip_atomic_fetch_add(&y[(size_t)(val >> 1) * DD + gcol], w * v,
                                   __ATOMIC_RELAXED, __HIP_MEMORY_SCOPE_AGENT);
          }
        }
      }
    }
  }
}

// ----------------------------------------------------------------- launch ---

extern "C" void kernel_launch(void* const* d_in, const int* in_sizes, int n_in,
                              void* d_out, int out_size, void* d_ws, size_t ws_size,
                              hipStream_t stream) {
  const float* x  = (const float*)d_in[0];
  const float* Wg = (const float*)d_in[1];
  const float* bg = (const float*)d_in[2];
  const float* W1 = (const float*)d_in[3];
  const float* b1 = (const float*)d_in[4];
  const float* W2 = (const float*)d_in[5];
  const float* b2 = (const float*)d_in[6];

  const int Ntok = in_sizes[0] / DD;              // 8192 tokens
  float* y        = (float*)d_out;                // [Ntok, D]
  float* gate_out = y + (size_t)Ntok * DD;        // [Ntok, E]

  const size_t nX = (size_t)Ntok * DD;
  const size_t nH = (size_t)Ntok * 2 * DD;
  const size_t nW = (size_t)EXPERTS * DD * DD;

  uint8_t* ws = (uint8_t*)d_ws;
  __bf16* hH  = (__bf16*)ws;                 ws += nH * 2;
  __bf16* hL  = (__bf16*)ws;                 ws += nH * 2;
  __bf16* xh  = (__bf16*)ws;                 ws += nX * 2;
  __bf16* xl  = (__bf16*)ws;                 ws += nX * 2;
  __bf16* w1h = (__bf16*)ws;                 ws += nW * 2;
  __bf16* w1l = (__bf16*)ws;                 ws += nW * 2;
  __bf16* w2h = (__bf16*)ws;                 ws += nW * 2;
  __bf16* w2l = (__bf16*)ws;                 ws += nW * 2;
  unsigned* cnt   = (unsigned*)ws;           ws += 256;
  unsigned* lists = (unsigned*)ws;           ws += (size_t)EXPERTS * Ntok * 4;
  float*    wlist = (float*)ws;

  hipMemsetAsync(y, 0, nX * sizeof(float), stream);
  hipMemsetAsync(cnt, 0, EXPERTS * sizeof(unsigned), stream);

  split_plain<<<(int)((nX / 4 + 255) / 256), 256, 0, stream>>>(x, xh, xl, (int)nX);
  dim3 tg(32, 32, EXPERTS);
  split_transpose<<<tg, 256, 0, stream>>>(W1, w1h, w1l);
  split_transpose<<<tg, 256, 0, stream>>>(W2, w2h, w2l);

  gate_topk<<<(Ntok + 7) / 8, 256, 0, stream>>>(x, Wg, bg, gate_out, cnt, lists,
                                                wlist, Ntok);

  dim3 grid(DD / BN, (Ntok + BM - 1) / BM, EXPERTS);
  moe_gemm<1><<<grid, 256, 0, stream>>>(xh, xl, w1h, w1l, b1, cnt, lists, wlist,
                                        hH, hL, y, Ntok);
  moe_gemm<2><<<grid, 256, 0, stream>>>(hH, hL, w2h, w2l, b2, cnt, lists, wlist,
                                        hH, hL, y, Ntok);
}
